// _AutoCorrelation_47193100648690
// MI455X (gfx1250) — matrix operand order repaired and checked by reading, not compiled, not run on hardware
//
#include <hip/hip_runtime.h>
#include <math.h>
#include <stdint.h>

typedef unsigned short u16;
typedef unsigned int   u32;
typedef __attribute__((ext_vector_type(16))) __bf16 v16bf;
typedef __attribute__((ext_vector_type(8)))  float  v8f;
typedef __attribute__((ext_vector_type(4)))  unsigned int u32x4;
typedef __attribute__((ext_vector_type(8)))  unsigned int u32x8;
typedef __attribute__((ext_vector_type(4)))  float  f32x4;

#define BATCH 16
#define SEQ   2048
#define DMODEL 512
#define TOPK  4

union Frag { v16bf v; u32x4 q[2]; };

static __device__ inline u16 f2bf(float f) {
    union { float f; u32 u; } x; x.f = f;
    u32 r = x.u + 0x7fffu + ((x.u >> 16) & 1u);   // round-to-nearest-even
    return (u16)(r >> 16);
}

static __device__ inline v8f wmma_bf16(v16bf a, v16bf b, v8f c) {
    return __builtin_amdgcn_wmma_f32_16x16x32_bf16(false, a, false, b, (short)0, c, false, false);
}

// Stage a 16-row x 512-col bf16 tile (16 KB) from global memory into LDS with the
// Tensor Data Mover. D# per CDNA5 ISA sec 8.3/8.4 (2-group form, VADDR2/3 = NULL):
//   group0: count=1 | lds_addr | global_addr[56:0] | type=2
//   group1: data_size=2B, tensor_dim0=512, tensor_dim1=16, tile=512x16, stride0=512
// Caller must be wave-uniform; follow with s_wait_tensorcnt + block barrier.
static __device__ inline void tdm_stage_tile(const u16* gsrc, u16* lds_dst) {
    union { u32x4 v; u32 d[4]; } g0;
    union { u32x8 v; u32 d[8]; } g1;
    unsigned long long ga = (unsigned long long)(uintptr_t)gsrc;
    u32 la = (u32)(uintptr_t)lds_dst;            // low 32 bits = LDS byte offset
    g0.d[0] = 1u;                                  // count = 1 valid descriptor
    g0.d[1] = la;                                  // lds_addr
    g0.d[2] = (u32)(ga & 0xffffffffu);             // global_addr[31:0]
    g0.d[3] = (u32)((ga >> 32) & 0x01ffffffu) | (2u << 30); // addr[56:32] | type=2
    g1.d[0] = (1u << 16);                          // data_size = 1 -> 2 bytes
    g1.d[1] = (u32)DMODEL << 16;                   // tensor_dim0[15:0] = 512
    g1.d[2] = (16u << 16);                         // dim0 hi=0 | tensor_dim1 = 16
    g1.d[3] = ((u32)DMODEL << 16);                 // dim1 hi=0 | tile_dim0 = 512
    g1.d[4] = 16u;                                 // tile_dim1 = 16, tile_dim2 = 0
    g1.d[5] = (u32)DMODEL;                         // tensor_dim0_stride = 512
    g1.d[6] = 0u;                                  // stride0 hi | stride1 lo
    g1.d[7] = 0u;
    asm volatile("tensor_load_to_lds %0, %1" :: "s"(g0.v), "s"(g1.v) : "memory");
}

// ---------------------------------------------------------------- utilities
__global__ void k_f32_to_bf16(const float* __restrict__ src, u16* __restrict__ dst, int n) {
    int i = blockIdx.x * blockDim.x + threadIdx.x;
    int stride = gridDim.x * blockDim.x;
    for (; i < n; i += stride) dst[i] = f2bf(src[i]);
}

__global__ void k_zero(float* __restrict__ p, int n) {
    int i = blockIdx.x * blockDim.x + threadIdx.x;
    if (i < n) p[i] = 0.f;
}

// ---------------------------------------------------------------- QKV GEMM
// Q/K/V[m,n] = sum_d x[m,d] * W[n,d] + b[n]
// Block = 8 waves sharing one m-tile (TDM-staged in LDS); each wave owns one n-tile.
__global__ __launch_bounds__(256) void k_qkv(const u16* __restrict__ xbf,
                                             const u16* __restrict__ wbf,   // Wq|Wk|Wv, (512,512) row-major (e,d)
                                             const float* __restrict__ bq,
                                             const float* __restrict__ bk,
                                             const float* __restrict__ bv,
                                             u16* __restrict__ qbf,
                                             u16* __restrict__ kbf,
                                             float* __restrict__ vf) {
    __shared__ u16 sA[16 * DMODEL];                // 16 KB staged x tile
    const int lane = threadIdx.x & 31;
    const int wv   = threadIdx.x >> 5;             // 0..7
    const int mtile = blockIdx.x >> 2;             // 0..2047 (4 blocks per m-tile)
    const int ntile = ((blockIdx.x & 3) << 3) + wv; // 0..31
    const int lh = lane & 15;
    const int hi = (lane >> 4) & 1;

    if (threadIdx.x < 32) {                        // wave 0 drives the TDM
        tdm_stage_tile(xbf + (size_t)mtile * 16 * DMODEL, sA);
        __builtin_amdgcn_s_wait_tensorcnt(0);
    }
    __syncthreads();

    // A fragments from LDS: 16x32 bf16 tiles, row M = lh per lane.
    const u16* arow = sA + (size_t)lh * DMODEL;
    const int alo = hi ? 16 : 0;
    Frag a[16];
#pragma unroll
    for (int kc = 0; kc < 16; ++kc) {
        const u16* p = arow + kc * 32 + alo;
        a[kc].q[0] = *(const u32x4*)(p);
        a[kc].q[1] = *(const u32x4*)(p + 8);
    }

    const int n   = ntile * 16 + lh;
    const int blo = hi ? 16 : 0;

    for (int mat = 0; mat < 3; ++mat) {
        const u16* wrow = wbf + (size_t)mat * DMODEL * DMODEL + (size_t)n * DMODEL + blo;
        // preload all 16 B fragments, then burst the WMMAs
        Frag bfr[16];
#pragma unroll
        for (int kc = 0; kc < 16; ++kc) {
            bfr[kc].q[0] = *(const u32x4*)(wrow + kc * 32);
            bfr[kc].q[1] = *(const u32x4*)(wrow + kc * 32 + 8);
        }
        v8f c = {0.f, 0.f, 0.f, 0.f, 0.f, 0.f, 0.f, 0.f};
#pragma unroll
        for (int kc = 0; kc < 16; ++kc)
            c = wmma_bf16(a[kc].v, bfr[kc].v, c);

        const float* bias = (mat == 0) ? bq : (mat == 1) ? bk : bv;
        const float bval = bias[n];
#pragma unroll
        for (int j = 0; j < 8; ++j) {
            const int M = mtile * 16 + j + hi * 8;
            const float r = c[j] + bval;
            const size_t off = (size_t)M * DMODEL + n;
            if (mat == 0)      qbf[off] = f2bf(r);
            else if (mat == 1) kbf[off] = f2bf(r);
            else               vf[off]  = r;
        }
    }
}

// ---------------------------------------------------------------- correlation
// mean_corr[b,tau] = (1/D) * sum_t sum_e Q[b,t,e] * K[b,(t-tau)%L,e]
// = diagonal sums of G = Q K^T. WMMA tiles -> LDS float atomics on tau -> global flush.
__global__ __launch_bounds__(256) void k_corr(const u16* __restrict__ qbf,
                                              const u16* __restrict__ kbf,
                                              float* __restrict__ mc) {
    __shared__ float smc[SEQ];                     // 8 KB tau accumulators
    __shared__ u16 sA[16 * DMODEL];                // 16 KB staged Q tile
    const int b    = blockIdx.x >> 7;              // 16 batches
    const int ti   = blockIdx.x & 127;             // 128 t-tiles
    const int lane = threadIdx.x & 31;
    const int wv   = threadIdx.x >> 5;             // 8 waves
    const int lh   = lane & 15;
    const int hi   = (lane >> 4) & 1;

    for (int i = threadIdx.x; i < SEQ; i += 256) smc[i] = 0.f;
    if (threadIdx.x < 32) {                        // wave 0 drives the TDM
        tdm_stage_tile(qbf + (size_t)(b * SEQ + ti * 16) * DMODEL, sA);
        __builtin_amdgcn_s_wait_tensorcnt(0);
    }
    __syncthreads();

    const u16* arow = sA + (size_t)lh * DMODEL;
    const int alo = hi ? 16 : 0;
    Frag a[16];
#pragma unroll
    for (int kc = 0; kc < 16; ++kc) {
        const u16* p = arow + kc * 32 + alo;
        a[kc].q[0] = *(const u32x4*)(p);
        a[kc].q[1] = *(const u32x4*)(p + 8);
    }

    const int blo = hi ? 16 : 0;

    for (int st = 0; st < 16; ++st) {
        const int si = wv * 16 + st;               // 0..127
        const int s  = si * 16 + lh;
        const u16* brow = kbf + (size_t)(b * SEQ + s) * DMODEL + blo;
        Frag bfr[16];
#pragma unroll
        for (int kc = 0; kc < 16; ++kc) {
            bfr[kc].q[0] = *(const u32x4*)(brow + kc * 32);
            bfr[kc].q[1] = *(const u32x4*)(brow + kc * 32 + 8);
        }
        v8f c = {0.f, 0.f, 0.f, 0.f, 0.f, 0.f, 0.f, 0.f};
#pragma unroll
        for (int kc = 0; kc < 16; ++kc)
            c = wmma_bf16(a[kc].v, bfr[kc].v, c);
#pragma unroll
        for (int j = 0; j < 8; ++j) {
            const int t   = ti * 16 + j + hi * 8;
            const int tau = (t - s) & (SEQ - 1);
            atomicAdd(&smc[tau], c[j]);
        }
    }
    __syncthreads();
    const float scale = 1.0f / (float)DMODEL;
    for (int i = threadIdx.x; i < SEQ; i += 256)
        atomicAdd(&mc[(size_t)b * SEQ + i], smc[i] * scale);
}

// ---------------------------------------------------------------- top-4 + softmax
__global__ void k_topk(const float* __restrict__ mc, int* __restrict__ topi,
                       float* __restrict__ topw) {
    const int b = threadIdx.x;
    if (b >= BATCH) return;
    const float* row = mc + (size_t)b * SEQ;
    float bv[TOPK] = {-3.4e38f, -3.4e38f, -3.4e38f, -3.4e38f};
    int   bi[TOPK] = {0, 0, 0, 0};
    for (int i = 0; i < SEQ; ++i) {
        const float v = row[i];
        if (v > bv[TOPK - 1]) {
            int k = TOPK - 1;
            while (k > 0 && v > bv[k - 1]) { bv[k] = bv[k - 1]; bi[k] = bi[k - 1]; --k; }
            bv[k] = v; bi[k] = i;
        }
    }
    const float mx = bv[0];
    float e[TOPK], s = 0.f;
    for (int k = 0; k < TOPK; ++k) { e[k] = __expf(bv[k] - mx); s += e[k]; }
    const float inv = 1.0f / s;
    for (int k = 0; k < TOPK; ++k) {
        topi[b * TOPK + k] = bi[k];
        topw[b * TOPK + k] = e[k] * inv;
    }
}

// ---------------------------------------------------------------- aggregation
// out[b,t,:] = sum_k w[b,k] * V[b,(t-idx_k)%L,:]
__global__ __launch_bounds__(128) void k_agg(const float* __restrict__ V,
                                             const int* __restrict__ topi,
                                             const float* __restrict__ topw,
                                             float* __restrict__ out) {
    const int bt = blockIdx.x;
    const int b  = bt >> 11;          // / 2048
    const int t  = bt & (SEQ - 1);
    const int d4 = threadIdx.x;       // 128 float4 per row

    f32x4 acc = {0.f, 0.f, 0.f, 0.f};
#pragma unroll
    for (int k = 0; k < TOPK; ++k) {
        const int   s = (t - topi[b * TOPK + k]) & (SEQ - 1);
        const float w = topw[b * TOPK + k];
        const f32x4 v = *(const f32x4*)(V + (size_t)(b * SEQ + s) * DMODEL + d4 * 4);
        acc += v * w;
    }
    *(f32x4*)(out + (size_t)(b * SEQ + t) * DMODEL + d4 * 4) = acc;
}

// ---------------------------------------------------------------- launcher
extern "C" void kernel_launch(void* const* d_in, const int* in_sizes, int n_in,
                              void* d_out, int out_size, void* d_ws, size_t ws_size,
                              hipStream_t stream) {
    const float* x  = (const float*)d_in[0];
    const float* Wq = (const float*)d_in[1];
    const float* bq = (const float*)d_in[2];
    const float* Wk = (const float*)d_in[3];
    const float* bk = (const float*)d_in[4];
    const float* Wv = (const float*)d_in[5];
    const float* bv = (const float*)d_in[6];
    float* out = (float*)d_out;
    char* ws = (char*)d_ws;

    const size_t nTok = (size_t)BATCH * SEQ;           // 32768
    const size_t nX   = nTok * DMODEL;                 // 16.7M
    const size_t nW   = (size_t)DMODEL * DMODEL;       // 262144

    size_t off = 0;
    u16* xbf = (u16*)(ws + off); off += nX * 2;        // x in bf16
    u16* wbf = (u16*)(ws + off); off += 3 * nW * 2;    // Wq|Wk|Wv in bf16
    u16* qbf = (u16*)(ws + off); off += nX * 2;        // Q bf16
    u16* kbf = (u16*)(ws + off); off += nX * 2;        // K bf16
    float* vf = (float*)(ws + off); off += nX * 4;     // V f32
    float* mc = (float*)(ws + off); off += nTok * 4;   // mean_corr (B,L)
    int*   topi = (int*)(ws + off); off += BATCH * TOPK * 4;
    float* topw = (float*)(ws + off); off += BATCH * TOPK * 4;

    // 1. convert to bf16 + zero accumulators
    k_f32_to_bf16<<<4096, 256, 0, stream>>>(x, xbf, (int)nX);
    k_f32_to_bf16<<<512, 256, 0, stream>>>(Wq, wbf + 0 * nW, (int)nW);
    k_f32_to_bf16<<<512, 256, 0, stream>>>(Wk, wbf + 1 * nW, (int)nW);
    k_f32_to_bf16<<<512, 256, 0, stream>>>(Wv, wbf + 2 * nW, (int)nW);
    k_zero<<<(int)(nTok / 256), 256, 0, stream>>>(mc, (int)nTok);

    // 2. QKV projections: 2048 m-tiles * 4 blocks each, 8 waves/block
    k_qkv<<<8192, 256, 0, stream>>>(xbf, wbf, bq, bk, bv, qbf, kbf, vf);

    // 3. circular auto-correlation via Gram-diagonal sums
    k_corr<<<BATCH * 128, 256, 0, stream>>>(qbf, kbf, mc);

    // 4. top-4 + softmax
    k_topk<<<1, 32, 0, stream>>>(mc, topi, topw);

    // 5. weighted circular gather of V
    k_agg<<<BATCH * SEQ, 128, 0, stream>>>(vf, topi, topw, out);
}
